// ContinuousGraphLossNet_90323162235498
// MI455X (gfx1250) — compile-verified, matrix operand order checked by reading, not run on hardware
//
#include <hip/hip_runtime.h>
#include <hip/hip_fp16.h>
#include <math.h>

typedef __attribute__((ext_vector_type(16))) _Float16 v16h;
typedef __attribute__((ext_vector_type(8)))  _Float16 v8h;
typedef __attribute__((ext_vector_type(8)))  float    v8f;

#define DEV __device__ __forceinline__

static constexpr int Bsz = 256, Nn = 133, T = 134, Dm = 256, Lly = 6, NH = 8;
static constexpr int MROWS = Bsz * T;   // 34304 = 268*128

DEV float gelu_exact(float x) { return 0.5f * x * (1.0f + erff(x * 0.70710678118654752f)); }

DEV v8f zero8() { v8f z = {0.f,0.f,0.f,0.f,0.f,0.f,0.f,0.f}; return z; }

// Async DMA: 16 bytes global -> LDS, tracked by ASYNCcnt (CDNA5 path).
// GV mode: per-lane 64-bit global address, per-lane 32-bit LDS address
// (low 32 bits of the flat pointer == LDS offset per the aperture rule).
DEV void async_cp16(void* lds, const void* g) {
  unsigned l32 = (unsigned)(unsigned long long)lds;
  asm volatile("global_load_async_to_lds_b128 %0, %1, off"
               :: "v"(l32), "v"(g) : "memory");
}
DEV void wait_async0() { asm volatile("s_wait_asynccnt 0" ::: "memory"); }

// Load a 16x32 f16 fragment from LDS (row-major, `stride` halves per row)
// following the CDNA5 WMMA A-operand lane layout:
//   lanes 0-15 : row = lane,    K = 0..7  and 16..23
//   lanes16-31 : row = lane-16, K = 8..15 and 24..31
// -> per lane: two contiguous 16B chunks.
DEV v16h ld_frag(const _Float16* p, int stride) {
  int lane = threadIdx.x & 31;
  const _Float16* q = p + (lane & 15) * stride + ((lane >> 4) << 3);
  v8h lo = *(const v8h*)q;
  v8h hi = *(const v8h*)(q + 16);
  return __builtin_shufflevector(lo, hi, 0,1,2,3,4,5,6,7,8,9,10,11,12,13,14,15);
}

DEV v8f wmma16(v16h a, v16h b, v8f c) {
  return __builtin_amdgcn_wmma_f32_16x16x32_f16(false, a, false, b, (short)0, c, false, false);
}

// ---------------------------------------------------------------------------
// One-time weight convert+transpose: W[l][k][n] (f32) -> wT[(g*L+l)][n][k] f16
// ---------------------------------------------------------------------------
__global__ void wconv_kernel(const float* __restrict__ wq, const float* __restrict__ wk,
                             const float* __restrict__ wv, const float* __restrict__ wo,
                             const float* __restrict__ w1, const float* __restrict__ w2,
                             _Float16* __restrict__ out) {
  int tid = threadIdx.x;
  int id  = blockIdx.x;
  int nrow = id & 255;
  int l    = (id >> 8) % Lly;
  int g    = id / (256 * Lly);
  const float* W = (g==0)?wq:(g==1)?wk:(g==2)?wv:(g==3)?wo:(g==4)?w1:w2;
  float val = W[(size_t)l*65536 + (size_t)tid*256 + nrow];
  out[((size_t)(g*Lly + l)*256 + nrow)*256 + tid] = (_Float16)val;
}

// ---------------------------------------------------------------------------
// Attention bias matrix (T x T) from adj/dist/deg — batch-independent.
// ---------------------------------------------------------------------------
__global__ void bias_kernel(const float* __restrict__ adj, const float* __restrict__ dist,
                            const float* __restrict__ noedge, const float* __restrict__ dscale,
                            const float* __restrict__ gma, const float* __restrict__ v2n,
                            const float* __restrict__ n2v, const float* __restrict__ vself,
                            float* __restrict__ biasb) {
  __shared__ float deg[Nn];
  __shared__ float dmax;
  int tid = threadIdx.x;
  if (tid < Nn) { float s = 0.f; for (int j = 0; j < Nn; ++j) s += adj[tid*Nn + j]; deg[tid] = s; }
  __syncthreads();
  if (tid == 0) { float m = 0.f; for (int i = 0; i < Nn; ++i) m = fmaxf(m, deg[i]); dmax = m; }
  __syncthreads();
  for (int idx = tid; idx < T*T; idx += 256) {
    int i = idx / T, j = idx % T;
    float val;
    if (i == 0 && j == 0)      val = vself[0];
    else if (i == 0)           val = v2n[0];
    else if (j == 0)           val = n2v[0];
    else {
      int a = i-1, c = j-1;
      float e = -gma[0] * dist[a*Nn + c];
      if (!(adj[a*Nn + c] > 0.f)) e += noedge[0];
      e += dscale[0] * (deg[a] + deg[c]) / (dmax + 1e-6f);
      val = e;
    }
    biasb[idx] = val;
  }
}

// ---------------------------------------------------------------------------
// Input embedding: z[b][0] = cls ; z[b][1+n] = concat(x2d,y3d,je) @ in_w + in_b
// ---------------------------------------------------------------------------
__global__ void embed_kernel(const float* __restrict__ x2d, const float* __restrict__ y3d,
                             const float* __restrict__ je,  const float* __restrict__ in_w,
                             const float* __restrict__ in_b, const float* __restrict__ cls,
                             float* __restrict__ z) {
  int tid = threadIdx.x;
  int row = blockIdx.x;
  int b = row / T, t = row % T;
  size_t o = (size_t)row * Dm + tid;
  if (t == 0) { z[o] = cls[tid]; return; }
  int n = t - 1;
  float acc = in_b[tid];
  acc += x2d[((size_t)b*Nn + n)*2 + 0] * in_w[0*Dm + tid];
  acc += x2d[((size_t)b*Nn + n)*2 + 1] * in_w[1*Dm + tid];
  acc += y3d[((size_t)b*Nn + n)*3 + 0] * in_w[2*Dm + tid];
  acc += y3d[((size_t)b*Nn + n)*3 + 1] * in_w[3*Dm + tid];
  acc += y3d[((size_t)b*Nn + n)*3 + 2] * in_w[4*Dm + tid];
#pragma unroll
  for (int j = 0; j < 32; ++j) acc += je[(size_t)n*32 + j] * in_w[(5 + j)*Dm + tid];
  z[o] = acc;
}

// ---------------------------------------------------------------------------
// LayerNorm over D=256 (one block/row), f32 in -> f16 out.
// row_mul=1: all rows; row_mul=T: cls rows only (final LN).
// ---------------------------------------------------------------------------
__global__ void ln_kernel(const float* __restrict__ z, const float* __restrict__ g,
                          const float* __restrict__ bb, _Float16* __restrict__ out,
                          int row_mul) {
  __shared__ float red[256];
  __shared__ float red2[256];
  int tid = threadIdx.x;
  size_t zrow = (size_t)blockIdx.x * row_mul;
  float x = z[zrow*Dm + tid];
  red[tid] = x; red2[tid] = x*x;
  __syncthreads();
  for (int s = 128; s > 0; s >>= 1) {
    if (tid < s) { red[tid] += red[tid+s]; red2[tid] += red2[tid+s]; }
    __syncthreads();
  }
  float mean = red[0] * (1.f/256.f);
  float var  = red2[0] * (1.f/256.f) - mean*mean;
  float r = rsqrtf(var + 1e-5f);
  out[(size_t)blockIdx.x*Dm + tid] = (_Float16)((x - mean) * r * g[tid] + bb[tid]);
}

// ---------------------------------------------------------------------------
// WMMA GEMM: C(M x 256) = A(M x 256, f16) * Bt^T + bias
//   Bt is f16, N-major (Bt[n][k]) so B-fragments load like A-fragments.
//   mode 0: store f16   mode 1: f32 += (residual)   mode 2: GELU -> f16
// Block: 256 thr / 8 waves; tile 128x64.  K=256 is small enough to stage
// the ENTIRE A (128x256) and B (64x256) panels in LDS with one round of
// async global->LDS DMA, then run a barrier-free ds_load -> v_wmma loop.
// ---------------------------------------------------------------------------
__global__ __launch_bounds__(256)
void gemm_kernel(const _Float16* __restrict__ A, const _Float16* __restrict__ Bt,
                 const float* __restrict__ bias, _Float16* __restrict__ outH,
                 float* __restrict__ outF, int mode) {
  __shared__ _Float16 As[128*264];   // 67,584 B (row stride 264 halves)
  __shared__ _Float16 Bs[64*264];    // 33,792 B
  const int tid = threadIdx.x, lane = tid & 31, wid = tid >> 5;
  const int bm = blockIdx.x >> 2, bn = blockIdx.x & 3;
  const int row0 = bm * 128, col0 = bn * 64;
  const int wm = wid & 3, wn = wid >> 2;

  // Stage full panels via async DMA (ASYNCcnt), 16B per issue.
  for (int c = tid; c < 128*32; c += 256) {        // A: 4096 chunks
    int r = c >> 5, o = (c & 31) * 8;
    async_cp16(&As[r*264 + o], &A[(size_t)(row0 + r)*256 + o]);
  }
  for (int c = tid; c < 64*32; c += 256) {         // B: 2048 chunks
    int r = c >> 5, o = (c & 31) * 8;
    async_cp16(&Bs[r*264 + o], &Bt[(size_t)(col0 + r)*256 + o]);
  }
  wait_async0();
  __syncthreads();

  v8f a00 = zero8(), a01 = zero8(), a10 = zero8(), a11 = zero8();
#pragma unroll
  for (int kt = 0; kt < 8; ++kt) {
    int k0 = kt * 32;
    v16h fa0 = ld_frag(&As[(wm*32     )*264 + k0], 264);
    v16h fa1 = ld_frag(&As[(wm*32 + 16)*264 + k0], 264);
    v16h fb0 = ld_frag(&Bs[(wn*32     )*264 + k0], 264);
    v16h fb1 = ld_frag(&Bs[(wn*32 + 16)*264 + k0], 264);
    a00 = wmma16(fa0, fb0, a00);
    a01 = wmma16(fa0, fb1, a01);
    a10 = wmma16(fa1, fb0, a10);
    a11 = wmma16(fa1, fb1, a11);
  }

  v8f accs[4] = {a00, a01, a10, a11};
#pragma unroll
  for (int t4 = 0; t4 < 4; ++t4) {
    int ti = t4 >> 1, tj = t4 & 1;
    int r0 = row0 + wm*32 + ti*16 + ((lane >> 4) << 3);
    int c  = col0 + wn*32 + tj*16 + (lane & 15);
    float bv = bias[c];
#pragma unroll
    for (int r = 0; r < 8; ++r) {
      float v = accs[t4][r] + bv;
      size_t idx = (size_t)(r0 + r)*256 + c;
      if (mode == 1)      outF[idx] += v;
      else if (mode == 2) outH[idx]  = (_Float16)gelu_exact(v);
      else                outH[idx]  = (_Float16)v;
    }
  }
}

// ---------------------------------------------------------------------------
// Fused attention, one workgroup per (batch, head).  T=134 padded to 144/160.
// Q/K staged by async DMA; QK^T via 81 WMMAs -> f32 scores in LDS -> rowwise
// softmax (denominator deferred) with exp-scores written f16 aliased over the
// score rows -> AV via 90 WMMAs -> f16 output / rowsum.
// ---------------------------------------------------------------------------
__global__ __launch_bounds__(256, 1)
void attn_kernel(const _Float16* __restrict__ q, const _Float16* __restrict__ k,
                 const _Float16* __restrict__ v, const float* __restrict__ bias,
                 _Float16* __restrict__ out) {
  __shared__ float    S [144*148];   // scores f32; aliased by exp-scores f16 (stride 296 halves)
  __shared__ _Float16 Qs[144*40];
  __shared__ _Float16 Ks[144*40];
  __shared__ _Float16 Vt[32*168];    // V transposed: Vt[d][t], t padded to 160
  __shared__ float    rsum[144];
  _Float16* At = (_Float16*)S;       // row stride 296 halves == 148 floats

  const int tid = threadIdx.x, lane = tid & 31, wid = tid >> 5;
  const int b = blockIdx.x >> 3, h = blockIdx.x & 7;
  const size_t base = ((size_t)b * T) * Dm + h * 32;

  // Zero-fill Q/K pad rows 134..143 (ds stores), then DMA valid rows async.
  for (int c = tid; c < 10*4; c += 256) {
    int r = T + (c >> 2), o = (c & 3) * 8;
    float4 zv = {0.f, 0.f, 0.f, 0.f};
    *(float4*)&Qs[r*40 + o] = zv;
    *(float4*)&Ks[r*40 + o] = zv;
  }
  for (int c = tid; c < T*4; c += 256) {
    int r = c >> 2, o = (c & 3) * 8;
    async_cp16(&Qs[r*40 + o], &q[base + (size_t)r*Dm + o]);
    async_cp16(&Ks[r*40 + o], &k[base + (size_t)r*Dm + o]);
  }
  // Stage V transposed (zero-padded t 134..159) — elementwise, stays scalar.
  for (int c = tid; c < 160*32; c += 256) {
    int t = c >> 5, d = c & 31;
    _Float16 val = (t < T) ? v[base + (size_t)t*Dm + d] : (_Float16)0.f;
    Vt[d*168 + t] = val;
  }
  wait_async0();
  __syncthreads();

  // scores = Q K^T : 9x9 tiles of 16x16, K-dim = dk = 32 (one WMMA each)
  for (int tile = wid; tile < 81; tile += 8) {
    int ti = tile / 9, tj = tile % 9;
    v16h fa = ld_frag(&Qs[ti*16*40], 40);
    v16h fb = ld_frag(&Ks[tj*16*40], 40);
    v8f s = zero8();
    s = wmma16(fa, fb, s);
    int r0 = ti*16 + ((lane >> 4) << 3);
    int cc = tj*16 + (lane & 15);
#pragma unroll
    for (int r = 0; r < 8; ++r) S[(r0 + r)*148 + cc] = s[r];
  }
  __syncthreads();

  // Row softmax (one thread per row); exp written f16 over the same row.
  if (tid < T) {
    const float scale = 0.17677669529663689f;   // 1/sqrt(32)
    float mx = -1e30f;
    for (int s = 0; s < T; ++s) {
      float x = S[tid*148 + s]*scale + bias[tid*T + s];
      mx = fmaxf(mx, x);
    }
    float sum = 0.f;
    for (int s = 0; s < T; ++s) {
      float x = S[tid*148 + s]*scale + bias[tid*T + s];
      float e = __expf(x - mx);
      sum += e;
      At[tid*296 + s] = (_Float16)e;     // write offset 2s trails read offset 4s
    }
    for (int s = T; s < 160; ++s) At[tid*296 + s] = (_Float16)0.f;
    rsum[tid] = sum;
  } else if (tid < 144) {
    for (int s = 0; s < 160; ++s) At[tid*296 + s] = (_Float16)0.f;
    rsum[tid] = 1.f;
  }
  __syncthreads();

  // out = attn @ V : 9x2 tiles of 16x16, K padded to 160 (5 chunks of 32)
  for (int tile = wid; tile < 18; tile += 8) {
    int tm = tile >> 1, tn = tile & 1;
    v8f o8 = zero8();
#pragma unroll
    for (int kc = 0; kc < 5; ++kc) {
      v16h fa = ld_frag(&At[tm*16*296 + kc*32], 296);
      v16h fb = ld_frag(&Vt[tn*16*168 + kc*32], 168);
      o8 = wmma16(fa, fb, o8);
    }
    int r0 = tm*16 + ((lane >> 4) << 3);
    int d  = tn*16 + (lane & 15);
#pragma unroll
    for (int r = 0; r < 8; ++r) {
      int t = r0 + r;
      if (t < T) out[base + (size_t)t*Dm + d] = (_Float16)(o8[r] / rsum[t]);
    }
  }
}

// ---------------------------------------------------------------------------
// Head: c(256) -> GELU(c@W1+b1) -> @W2 + b2 ; one block per batch element.
// ---------------------------------------------------------------------------
__global__ void head_kernel(const _Float16* __restrict__ xlnf, const float* __restrict__ w1,
                            const float* __restrict__ b1, const float* __restrict__ w2,
                            const float* __restrict__ b2, float* __restrict__ out) {
  __shared__ float cbuf[256];
  __shared__ float red[256];
  int tid = threadIdx.x, b = blockIdx.x;
  cbuf[tid] = (float)xlnf[(size_t)b*256 + tid];
  __syncthreads();
  float acc = b1[tid];
  for (int i = 0; i < 256; ++i) acc += cbuf[i] * w1[(size_t)i*256 + tid];
  red[tid] = gelu_exact(acc) * w2[tid];
  __syncthreads();
  for (int s = 128; s > 0; s >>= 1) {
    if (tid < s) red[tid] += red[tid + s];
    __syncthreads();
  }
  if (tid == 0) out[b] = red[0] + b2[0];
}

// ---------------------------------------------------------------------------
extern "C" void kernel_launch(void* const* d_in, const int* in_sizes, int n_in,
                              void* d_out, int out_size, void* d_ws, size_t ws_size,
                              hipStream_t stream) {
  (void)in_sizes; (void)n_in; (void)out_size; (void)ws_size;
  const float* x2d    = (const float*)d_in[0];
  const float* y3d    = (const float*)d_in[1];
  const float* je     = (const float*)d_in[2];
  const float* in_w   = (const float*)d_in[3];
  const float* in_b   = (const float*)d_in[4];
  const float* cls    = (const float*)d_in[5];
  const float* noedge = (const float*)d_in[6];
  const float* dscale = (const float*)d_in[7];
  const float* gma    = (const float*)d_in[8];
  const float* v2n    = (const float*)d_in[9];
  const float* n2v    = (const float*)d_in[10];
  const float* vself  = (const float*)d_in[11];
  const float* ln1g   = (const float*)d_in[12];
  const float* ln1b   = (const float*)d_in[13];
  const float* wq     = (const float*)d_in[14];
  const float* bq     = (const float*)d_in[15];
  const float* wk     = (const float*)d_in[16];
  const float* bk     = (const float*)d_in[17];
  const float* wv     = (const float*)d_in[18];
  const float* bv     = (const float*)d_in[19];
  const float* wo     = (const float*)d_in[20];
  const float* bo     = (const float*)d_in[21];
  const float* ln2g   = (const float*)d_in[22];
  const float* ln2b   = (const float*)d_in[23];
  const float* mw1    = (const float*)d_in[24];
  const float* mb1    = (const float*)d_in[25];
  const float* mw2    = (const float*)d_in[26];
  const float* mb2    = (const float*)d_in[27];
  const float* lnfg   = (const float*)d_in[28];
  const float* lnfb   = (const float*)d_in[29];
  const float* hw1    = (const float*)d_in[30];
  const float* hb1    = (const float*)d_in[31];
  const float* hw2    = (const float*)d_in[32];
  const float* hb2    = (const float*)d_in[33];
  const float* adj    = (const float*)d_in[34];
  const float* dist   = (const float*)d_in[35];

  // Workspace carve-up (256B aligned)
  char* w = (char*)d_ws;
  size_t off = 0;
  auto carve = [&](size_t bytes) -> char* {
    char* p = w + off;
    off = (off + bytes + 255) & ~(size_t)255;
    return p;
  };
  float*    z     = (float*)   carve((size_t)MROWS * Dm * 4);
  _Float16* xln   = (_Float16*)carve((size_t)MROWS * Dm * 2);
  _Float16* qb    = (_Float16*)carve((size_t)MROWS * Dm * 2);
  _Float16* kb2   = (_Float16*)carve((size_t)MROWS * Dm * 2);
  _Float16* vb2   = (_Float16*)carve((size_t)MROWS * Dm * 2);
  _Float16* ao    = (_Float16*)carve((size_t)MROWS * Dm * 2);
  _Float16* hb    = (_Float16*)carve((size_t)MROWS * Dm * 2);
  _Float16* wT    = (_Float16*)carve((size_t)6 * Lly * 65536 * 2);
  float*    biasb = (float*)   carve((size_t)T * T * 4);
  _Float16* xlnf  = (_Float16*)carve((size_t)Bsz * Dm * 2);

  wconv_kernel<<<6*Lly*256, 256, 0, stream>>>(wq, wk, wv, wo, mw1, mw2, wT);
  bias_kernel <<<1,         256, 0, stream>>>(adj, dist, noedge, dscale, gma, v2n, n2v, vself, biasb);
  embed_kernel<<<MROWS,     256, 0, stream>>>(x2d, y3d, je, in_w, in_b, cls, z);

  const int GB = (MROWS/128) * 4;   // 1072 blocks
  for (int l = 0; l < Lly; ++l) {
    ln_kernel  <<<MROWS, 256, 0, stream>>>(z, ln1g + l*Dm, ln1b + l*Dm, xln, 1);
    gemm_kernel<<<GB,    256, 0, stream>>>(xln, wT + (size_t)(0*Lly + l)*65536, bq + l*Dm, qb,  nullptr, 0);
    gemm_kernel<<<GB,    256, 0, stream>>>(xln, wT + (size_t)(1*Lly + l)*65536, bk + l*Dm, kb2, nullptr, 0);
    gemm_kernel<<<GB,    256, 0, stream>>>(xln, wT + (size_t)(2*Lly + l)*65536, bv + l*Dm, vb2, nullptr, 0);
    attn_kernel<<<Bsz*NH, 256, 0, stream>>>(qb, kb2, vb2, biasb, ao);
    gemm_kernel<<<GB,    256, 0, stream>>>(ao,  wT + (size_t)(3*Lly + l)*65536, bo + l*Dm, nullptr, z, 1);
    ln_kernel  <<<MROWS, 256, 0, stream>>>(z, ln2g + l*Dm, ln2b + l*Dm, xln, 1);
    gemm_kernel<<<GB,    256, 0, stream>>>(xln, wT + (size_t)(4*Lly + l)*65536, mb1 + l*Dm, hb, nullptr, 2);
    gemm_kernel<<<GB,    256, 0, stream>>>(hb,  wT + (size_t)(5*Lly + l)*65536, mb2 + l*Dm, nullptr, z, 1);
  }

  ln_kernel  <<<Bsz, 256, 0, stream>>>(z, lnfg, lnfb, xlnf, T);
  head_kernel<<<Bsz, 256, 0, stream>>>(xlnf, hw1, hb1, hw2, hb2, (float*)d_out);
}
